// LSSTransform_36481452212257
// MI455X (gfx1250) — matrix operand order repaired and checked by reading, not compiled, not run on hardware
//
#include <hip/hip_runtime.h>

typedef __attribute__((ext_vector_type(16))) _Float16 v16h;
typedef __attribute__((ext_vector_type(8)))  _Float16 v8h;
typedef __attribute__((ext_vector_type(8)))  float    v8f;

// ---- problem constants (from reference) ----
#define BB      4
#define NNC     6
#define CINC    512
#define FHC     16
#define FWC     44
#define PIXC    (FHC*FWC)      // 704
#define DPTHC   59
#define CTC     64
#define MROWS   (DPTHC+CTC)    // 123
#define MPAD    128
#define NXG     128
#define NYG     128
#define TILE_PX 32
#define NTILES  (PIXC/TILE_PX) // 22
#define SXT_STR 528            // halfs per pixel row: 512 + 16 pad -> 1056B rows (32B aligned)
#define SY_STR  33             // floats per channel row (conflict-free: 33 % 64 coprime)

__global__ void lss_zero_kernel(float* __restrict__ p, int n) {
    int i = blockIdx.x * blockDim.x + threadIdx.x;
    if (i < n) p[i] = 0.0f;
}

__global__ __launch_bounds__(256)
void lss_fused_kernel(const float* __restrict__ x,
                      const float* __restrict__ rots,
                      const float* __restrict__ trans,
                      const float* __restrict__ intr,
                      const float* __restrict__ W,
                      const float* __restrict__ bias,
                      float* __restrict__ out) {
    __shared__ __attribute__((aligned(32))) _Float16 sXT[TILE_PX * SXT_STR]; // 33792 B
    __shared__ float sY[MPAD * SY_STR];                                      // 16896 B
    __shared__ float sBias[MPAD];
    __shared__ float sGeo[12];   // combine[9], trans[3]

    const int tid     = threadIdx.x;
    const int wgid    = blockIdx.x;
    const int tileIdx = wgid % NTILES;
    const int bn      = wgid / NTILES;
    const int bidx    = bn / NNC;
    const int px0     = tileIdx * TILE_PX;

    if (tid < MPAD) sBias[tid] = (tid < MROWS) ? bias[tid] : 0.0f;
    if (tid == 0) {
        const float* Km = intr + bn * 9;
        const float* Rm = rots + bn * 9;
        float a00=Km[0],a01=Km[1],a02=Km[2];
        float a10=Km[3],a11=Km[4],a12=Km[5];
        float a20=Km[6],a21=Km[7],a22=Km[8];
        float det = a00*(a11*a22-a12*a21) - a01*(a10*a22-a12*a20) + a02*(a10*a21-a11*a20);
        float id  = 1.0f / det;
        float inv[9] = {
            (a11*a22-a12*a21)*id, (a02*a21-a01*a22)*id, (a01*a12-a02*a11)*id,
            (a12*a20-a10*a22)*id, (a00*a22-a02*a20)*id, (a02*a10-a00*a12)*id,
            (a10*a21-a11*a20)*id, (a01*a20-a00*a21)*id, (a00*a11-a01*a10)*id };
        #pragma unroll
        for (int i = 0; i < 3; ++i)
            #pragma unroll
            for (int j = 0; j < 3; ++j)
                sGeo[i*3+j] = Rm[i*3+0]*inv[0*3+j] + Rm[i*3+1]*inv[1*3+j] + Rm[i*3+2]*inv[2*3+j];
        sGeo[9]  = trans[bn*3 + 0];
        sGeo[10] = trans[bn*3 + 1];
        sGeo[11] = trans[bn*3 + 2];
    }

    // ---- stage x tile (512 x 32 f32) -> LDS transposed f16 [pixel][channel] ----
    const float* xp = x + (size_t)bn * CINC * PIXC + px0;
    for (int i = tid; i < CINC * TILE_PX; i += 256) {
        int c = i >> 5;              // / TILE_PX
        int j = i & (TILE_PX - 1);
        sXT[j * SXT_STR + c] = (_Float16)xp[(size_t)c * PIXC + j];
    }
    __syncthreads();

    // ---- GEMM: 128x32 output tile, 8 waves x (16-row M stripe, 2 N-tiles) ----
    const int wave = tid >> 5;
    const int lane = tid & 31;
    const int l15  = lane & 15;
    const int g    = lane >> 4;
    const int m0   = wave * 16;
    int mrow = m0 + l15; if (mrow > MROWS - 1) mrow = MROWS - 1;  // pad rows duplicate, never read
    const float* Wrow = W + (size_t)mrow * CINC;

    v8f acc0 = {};
    v8f acc1 = {};
    for (int k0 = 0; k0 < CINC; k0 += 32) {
        // A fragment: 16-bit A 16x32 layout (V0-3: K=8g+2v(+1); V4-7: K=16+8g+2(v-4)(+1))
        union { v16h v; _Float16 h[16]; } A;
        #pragma unroll
        for (int vv = 0; vv < 8; ++vv) {
            int kk = (vv < 4) ? (8*g + 2*vv) : (16 + 8*g + 2*(vv - 4));
            float2 w2 = *(const float2*)(Wrow + k0 + kk);
            A.h[2*vv]   = (_Float16)w2.x;
            A.h[2*vv+1] = (_Float16)w2.y;
        }
        // B fragments: 16-bit B 32x16 layout (lane group g holds K = 8g + hh, then K = 16 + 8g + hh)
        union { v16h v; v8h p[2]; } B0, B1;
        const _Float16* bp0 = &sXT[(0  + l15) * SXT_STR + k0 + 8*g];
        B0.p[0] = *(const v8h*)(bp0);
        B0.p[1] = *(const v8h*)(bp0 + 16);
        const _Float16* bp1 = &sXT[(16 + l15) * SXT_STR + k0 + 8*g];
        B1.p[0] = *(const v8h*)(bp1);
        B1.p[1] = *(const v8h*)(bp1 + 16);

        acc0 = __builtin_amdgcn_wmma_f32_16x16x32_f16(false, A.v, false, B0.v, (short)0, acc0, false, false);
        acc1 = __builtin_amdgcn_wmma_f32_16x16x32_f16(false, A.v, false, B1.v, (short)0, acc1, false, false);
    }

    // ---- spill y tile to LDS (C/D layout: VGPR r -> row m0 + r + 8g, col = lane&15) ----
    {
        const int rowBase = m0 + 8 * g;
        #pragma unroll
        for (int r = 0; r < 8; ++r) {
            sY[(rowBase + r) * SY_STR + l15]      = acc0[r];
            sY[(rowBase + r) * SY_STR + 16 + l15] = acc1[r];
        }
    }
    __syncthreads();

    // ---- softmax over 59 depth logits per pixel (bias added here) ----
    if (tid < TILE_PX) {
        const int j = tid;
        float mx = -3.0e38f;
        for (int d = 0; d < DPTHC; ++d) {
            float v = sY[d * SY_STR + j] + sBias[d];
            mx = fmaxf(mx, v);
        }
        float sum = 0.0f;
        for (int d = 0; d < DPTHC; ++d) {
            float e = __expf(sY[d * SY_STR + j] + sBias[d] - mx);
            sY[d * SY_STR + j] = e;
            sum += e;
        }
        float inv = 1.0f / sum;
        for (int d = 0; d < DPTHC; ++d) sY[d * SY_STR + j] *= inv;
    }
    __syncthreads();

    // ---- geometry + voxel scatter: one (pixel, depth) pair per wave iteration ----
    const float c00 = sGeo[0], c01 = sGeo[1], c02 = sGeo[2];
    const float c10 = sGeo[3], c11 = sGeo[4], c12 = sGeo[5];
    const float c20 = sGeo[6], c21 = sGeo[7], c22 = sGeo[8];
    const float t0  = sGeo[9], t1  = sGeo[10], t2 = sGeo[11];
    float* outB = out + (size_t)bidx * CTC * NXG * NYG;

    for (int pp = wave; pp < TILE_PX * DPTHC; pp += 8) {
        const int j = pp & (TILE_PX - 1);
        const int d = pp >> 5;                 // TILE_PX == 32
        const int p = px0 + j;
        const int h = p / FWC;
        const int w = p - h * FWC;
        const float dsv = 1.0f + (float)d;     // arange(1, 60, 1)
        const float xs  = (float)w * (703.0f / 43.0f);  // linspace(0, 703, 44)
        const float ys  = (float)h * 17.0f;             // linspace(0, 255, 16)
        const float px_ = xs * dsv, py_ = ys * dsv;
        const float gx = c00*px_ + c01*py_ + c02*dsv + t0;
        const float gy = c10*px_ + c11*py_ + c12*dsv + t1;
        const float gz = c20*px_ + c21*py_ + c22*dsv + t2;
        const int ix = (int)floorf((gx + 51.2f) * 1.25f);
        const int iy = (int)floorf((gy + 51.2f) * 1.25f);
        const int iz = (int)floorf((gz + 10.0f) * 0.05f);
        if (ix >= 0 && ix < NXG && iy >= 0 && iy < NYG && iz == 0) {   // wave-uniform branch
            const float prob = sY[d * SY_STR + j];
            float* dst = outB + (size_t)ix * NYG + iy;
            int c = lane;
            float f0 = (sY[(DPTHC + c) * SY_STR + j] + sBias[DPTHC + c]) * prob;
            atomicAdd(dst + (size_t)c * (NXG * NYG), f0);
            c += 32;
            float f1 = (sY[(DPTHC + c) * SY_STR + j] + sBias[DPTHC + c]) * prob;
            atomicAdd(dst + (size_t)c * (NXG * NYG), f1);
        }
    }
}

extern "C" void kernel_launch(void* const* d_in, const int* in_sizes, int n_in,
                              void* d_out, int out_size, void* d_ws, size_t ws_size,
                              hipStream_t stream) {
    const float* x     = (const float*)d_in[0];
    const float* rots  = (const float*)d_in[1];
    const float* trans = (const float*)d_in[2];
    const float* intr  = (const float*)d_in[3];
    const float* W     = (const float*)d_in[4];
    const float* bias  = (const float*)d_in[5];
    float* out = (float*)d_out;

    lss_zero_kernel<<<(out_size + 255) / 256, 256, 0, stream>>>(out, out_size);
    lss_fused_kernel<<<BB * NNC * NTILES, 256, 0, stream>>>(x, rots, trans, intr, W, bias, out);
}
